// MultiHeadAttention_30039001269009
// MI455X (gfx1250) — matrix operand order repaired and checked by reading, not compiled, not run on hardware
//
#include <hip/hip_runtime.h>
#include <hip/hip_bf16.h>

// ---------------------------------------------------------------------------
// MI455X (gfx1250) multi-head attention, bf16 WMMA pipeline.
// B=4, S=2048, D=1024, H=16, HD=64.  All matmuls via v_wmma_f32_16x16x32_bf16.
// GEMMs double-buffer their LDS tiles; fp32->bf16 uses native HW converts.
// ---------------------------------------------------------------------------

typedef __attribute__((ext_vector_type(16))) __bf16 v16bf;
typedef __attribute__((ext_vector_type(8)))  __bf16 v8bf;
typedef __attribute__((ext_vector_type(4)))  __bf16 v4bf;
typedef __attribute__((ext_vector_type(8)))  float  v8f;
typedef __attribute__((ext_vector_type(4)))  float  v4f;

constexpr int Bc  = 4;
constexpr int Sc  = 2048;
constexpr int Dc  = 1024;
constexpr int Hc  = 16;
constexpr int HDc = 64;

// Native conversion: lowers to hardware bf16 cvt ops on gfx1250 (RNE).
__device__ __forceinline__ __bf16 f2bf(float f) { return (__bf16)f; }

// Build a 16-element bf16 fragment from two contiguous 8-element chunks.
__device__ __forceinline__ v16bf make_v16(const __bf16* p0, const __bf16* p1) {
  v8bf lo = *(const v8bf*)p0;
  v8bf hi = *(const v8bf*)p1;
  v16bf r;
#pragma unroll
  for (int i = 0; i < 8; ++i) { r[i] = lo[i]; r[i + 8] = hi[i]; }
  return r;
}

__device__ __forceinline__ v8f wmma_bf16(v16bf a, v16bf b, v8f c) {
  return __builtin_amdgcn_wmma_f32_16x16x32_bf16(false, a, false, b,
                                                 (short)0, c, false, false);
}

// ---------------------------------------------------------------------------
// GEMM: C[M,N] = A[M,K] * W[K,N] + bias.
//   A: fp32 (A_BF16=false) or bf16 (A_BF16=true), row-major.
//   W: fp32 row-major, converted to bf16 during LDS staging (transposed).
//   OUT_HEADS=true : write bf16 into [B*H, S, HD] head-split layout.
//   OUT_HEADS=false: write fp32 [M,N].
// Block tile 128x128, 8 waves, wave tile 64x32 (4x2 WMMA tiles), K-step 32,
// double-buffered LDS (one barrier per K-step, loads overlap WMMA).
// ---------------------------------------------------------------------------
template <bool A_BF16, bool OUT_HEADS>
__global__ __launch_bounds__(256)
void gemm_bf16_wmma(const void* __restrict__ Av, const float* __restrict__ W,
                    const float* __restrict__ bias, void* __restrict__ outv,
                    int M, int N, int K) {
  __shared__ __align__(16) __bf16 sA[2][128 * 32];   // [m][k]
  __shared__ __align__(16) __bf16 sB[2][128 * 32];   // [n][k] (transposed)

  const int tid  = threadIdx.x;
  const int lane = tid & 31;
  const int wave = tid >> 5;
  const int wm   = wave >> 2;          // 0..1
  const int wn   = wave & 3;           // 0..3
  const int hl   = lane >> 4;          // half-wave select
  const int lr   = lane & 15;
  const int m0   = blockIdx.x * 128;
  const int n0   = blockIdx.y * 128;

  const float*  Af = (const float*)Av;
  const __bf16* Ab = (const __bf16*)Av;

  // staging thread mapping (fixed per thread)
  const int arow = tid >> 3;              // 0..31
  const int acol = (tid & 7) * 4;         // 0,4,...,28
  const int bk   = tid >> 5;              // 0..7
  const int bn   = (tid & 31) * 4;        // 0..124

  auto stage = [&](int k0, int buf) {
    // A tile (128 x 32) -> sA[buf][m][k]
#pragma unroll
    for (int i = 0; i < 4; ++i) {
      const int r = arow + 32 * i;
      if (A_BF16) {
        const __bf16* p = Ab + (size_t)(m0 + r) * K + k0 + acol;
        *(v4bf*)(&sA[buf][r * 32 + acol]) = *(const v4bf*)p;
      } else {
        const float* p = Af + (size_t)(m0 + r) * K + k0 + acol;
        v4f a = *(const v4f*)p;
        sA[buf][r * 32 + acol + 0] = f2bf(a.x);
        sA[buf][r * 32 + acol + 1] = f2bf(a.y);
        sA[buf][r * 32 + acol + 2] = f2bf(a.z);
        sA[buf][r * 32 + acol + 3] = f2bf(a.w);
      }
    }
    // W tile (32 x 128), transposed -> sB[buf][n][k]
#pragma unroll
    for (int i = 0; i < 4; ++i) {
      const int k = bk + 8 * i;
      const float* p = W + (size_t)(k0 + k) * N + n0 + bn;
      v4f wv = *(const v4f*)p;
      sB[buf][(bn + 0) * 32 + k] = f2bf(wv.x);
      sB[buf][(bn + 1) * 32 + k] = f2bf(wv.y);
      sB[buf][(bn + 2) * 32 + k] = f2bf(wv.z);
      sB[buf][(bn + 3) * 32 + k] = f2bf(wv.w);
    }
    // prefetch tile-after-next into caches (global_prefetch_b8 path)
    if (k0 + 64 < K) {
      if (A_BF16)
        __builtin_prefetch(Ab + (size_t)(m0 + arow) * K + k0 + 64 + acol, 0, 1);
      else
        __builtin_prefetch(Af + (size_t)(m0 + arow) * K + k0 + 64 + acol, 0, 1);
      __builtin_prefetch(W + (size_t)(k0 + 64 + bk) * N + n0 + bn, 0, 1);
    }
  };

  v8f acc[4][2];
#pragma unroll
  for (int i = 0; i < 4; ++i)
#pragma unroll
    for (int j = 0; j < 2; ++j) acc[i][j] = v8f{};

  stage(0, 0);

  int buf = 0;
  for (int k0 = 0; k0 < K; k0 += 32, buf ^= 1) {
    __syncthreads();
    if (k0 + 32 < K) stage(k0 + 32, buf ^ 1);   // overlap next-tile loads

    v16bf afr[4];
#pragma unroll
    for (int i = 0; i < 4; ++i) {
      const __bf16* rp = &sA[buf][(wm * 64 + i * 16 + lr) * 32];
      afr[i] = make_v16(rp + (hl ? 16 : 0), rp + (hl ? 24 : 8));
    }
    v16bf bfr[2];
#pragma unroll
    for (int j = 0; j < 2; ++j) {
      const __bf16* cp = &sB[buf][(wn * 32 + j * 16 + lr) * 32 + hl * 16];
      bfr[j] = make_v16(cp, cp + 8);
    }
#pragma unroll
    for (int i = 0; i < 4; ++i)
#pragma unroll
      for (int j = 0; j < 2; ++j)
        acc[i][j] = wmma_bf16(afr[i], bfr[j], acc[i][j]);
  }

  // ---- epilogue: +bias, store ----
#pragma unroll
  for (int i = 0; i < 4; ++i) {
#pragma unroll
    for (int j = 0; j < 2; ++j) {
      const int n  = n0 + wn * 32 + j * 16 + lr;
      const float bv = bias[n];
#pragma unroll
      for (int r = 0; r < 8; ++r) {
        const int m = m0 + wm * 64 + i * 16 + r + hl * 8;
        const float v = acc[i][j][r] + bv;
        if (OUT_HEADS) {
          const int b = m / Sc, s = m % Sc, h = n / HDc, hd = n % HDc;
          ((__bf16*)outv)[((size_t)(b * Hc + h) * Sc + s) * HDc + hd] = f2bf(v);
        } else {
          ((float*)outv)[(size_t)m * N + n] = v;
        }
      }
    }
  }
}

// ---------------------------------------------------------------------------
// Flash attention over [B*H, S, HD] bf16 Q/K/V.
// Grid: (S/128, B*H). 8 waves/block; each wave owns 16 queries, O=16x64 in
// regs (4 v8f), online softmax (running max/sum per row via __shfl_xor).
// Per 32-key step: 4 QK^T WMMAs + 4 PV WMMAs.
// Output: bf16 [B, S, D] (heads merged) for the final projection.
// ---------------------------------------------------------------------------
__global__ __launch_bounds__(256)
void flash_attn(const __bf16* __restrict__ Q, const __bf16* __restrict__ Kg,
                const __bf16* __restrict__ V, const unsigned char* __restrict__ mask,
                __bf16* __restrict__ O) {
  __shared__ __align__(16) __bf16 sV[HDc * 32];       // [hd][key] transposed
  __shared__ __align__(16) __bf16 sP[8 * 16 * 32];    // per-wave P tile [q][key]

  const int tid  = threadIdx.x;
  const int lane = tid & 31;
  const int wave = tid >> 5;
  const int hl   = lane >> 4;
  const int lr   = lane & 15;

  const int q0 = blockIdx.x * 128;
  const int bh = blockIdx.y;
  const int b  = bh >> 4;                       // bh / H
  const size_t base = (size_t)bh * Sc * HDc;
  const __bf16* Qb = Q + base;
  const __bf16* Kb = Kg + base;
  const __bf16* Vb = V + base;
  const unsigned char* mb = mask + (size_t)b * Sc;
  const int qw = q0 + wave * 16;

  // Q fragments (16 queries x 64 hd => two 16x32 A fragments), loaded once.
  v16bf aQ[2];
#pragma unroll
  for (int f = 0; f < 2; ++f) {
    const __bf16* rp = Qb + (size_t)(qw + lr) * HDc + f * 32 + (hl ? 16 : 0);
    aQ[f] = make_v16(rp, rp + 8);
  }

  v8f o[4];
#pragma unroll
  for (int j = 0; j < 4; ++j) o[j] = v8f{};
  float mrow[8], lrow[8];
#pragma unroll
  for (int r = 0; r < 8; ++r) { mrow[r] = -__builtin_inff(); lrow[r] = 0.f; }

  __bf16* sPw = sP + wave * 16 * 32;

  for (int kc = 0; kc < Sc; kc += 32) {
    __syncthreads();
    // stage V chunk (32 keys x 64 hd) transposed into sV[hd][key]
    {
      const int key = tid >> 3;
      const int hg  = (tid & 7) * 8;
      v8bf vv = *(const v8bf*)(Vb + (size_t)(kc + key) * HDc + hg);
#pragma unroll
      for (int i = 0; i < 8; ++i) sV[(hg + i) * 32 + key] = vv[i];
      if (kc + 32 < Sc)
        __builtin_prefetch(Vb + (size_t)(kc + 32 + key) * HDc + hg, 0, 1);
    }
    __syncthreads();

    // QK^T: scores for 16 queries x 32 keys (two 16x16 C tiles)
    v8f c0 = v8f{}, c1 = v8f{};
#pragma unroll
    for (int f = 0; f < 2; ++f) {
      const __bf16* kp0 = Kb + (size_t)(kc + lr) * HDc + f * 32 + (hl ? 16 : 0);
      c0 = wmma_bf16(aQ[f], make_v16(kp0, kp0 + 8), c0);
      const __bf16* kp1 = Kb + (size_t)(kc + 16 + lr) * HDc + f * 32 + (hl ? 16 : 0);
      c1 = wmma_bf16(aQ[f], make_v16(kp1, kp1 + 8), c1);
    }

    const bool mk0 = mb[kc + lr] != 0;
    const bool mk1 = mb[kc + 16 + lr] != 0;

    // online softmax, per accumulator row r (row q = r + 8*hl)
#pragma unroll
    for (int r = 0; r < 8; ++r) {
      float s0 = mk0 ? c0[r] * 0.125f : -1e30f;   // 1/sqrt(64) = 0.125
      float s1 = mk1 ? c1[r] * 0.125f : -1e30f;
      float cm = fmaxf(s0, s1);
#pragma unroll
      for (int off = 1; off < 16; off <<= 1)
        cm = fmaxf(cm, __shfl_xor(cm, off, 32));
      const float mnew  = fmaxf(mrow[r], cm);
      const float scale = __expf(mrow[r] - mnew);
      const float p0 = __expf(s0 - mnew);
      const float p1 = __expf(s1 - mnew);
      float rs = p0 + p1;
#pragma unroll
      for (int off = 1; off < 16; off <<= 1)
        rs += __shfl_xor(rs, off, 32);
      lrow[r] = lrow[r] * scale + rs;
      mrow[r] = mnew;
#pragma unroll
      for (int j = 0; j < 4; ++j) o[j][r] *= scale;
      // C-layout -> [q][key] in wave-private LDS slab
      sPw[(r + hl * 8) * 32 + lr]      = f2bf(p0);
      sPw[(r + hl * 8) * 32 + 16 + lr] = f2bf(p1);
    }

    // P(16x32) @ V(32x64): A fragment from sPw, B fragments from sV
    const __bf16* pr = sPw + lr * 32 + (hl ? 16 : 0);
    v16bf aP = make_v16(pr, pr + 8);
#pragma unroll
    for (int j = 0; j < 4; ++j) {
      const __bf16* vp = sV + (j * 16 + lr) * 32 + (hl ? 16 : 0);
      o[j] = wmma_bf16(aP, make_v16(vp, vp + 8), o[j]);
    }
  }

  // normalize and store O as bf16 [B, S, D] (heads merged)
  const int h = bh & 15;
#pragma unroll
  for (int r = 0; r < 8; ++r) {
    const float inv = lrow[r] > 0.f ? 1.f / lrow[r] : 0.f;
    const int q = qw + r + hl * 8;
#pragma unroll
    for (int j = 0; j < 4; ++j) {
      const int col = h * HDc + j * 16 + lr;
      O[(size_t)(b * Sc + q) * Dc + col] = f2bf(o[j][r] * inv);
    }
  }
}

// ---------------------------------------------------------------------------
// Launcher
// ---------------------------------------------------------------------------
extern "C" void kernel_launch(void* const* d_in, const int* in_sizes, int n_in,
                              void* d_out, int out_size, void* d_ws, size_t ws_size,
                              hipStream_t stream) {
  (void)in_sizes; (void)n_in; (void)out_size; (void)ws_size;

  const float*         x    = (const float*)d_in[0];
  const unsigned char* mask = (const unsigned char*)d_in[1];
  const float* Wq = (const float*)d_in[2];
  const float* bq = (const float*)d_in[3];
  const float* Wk = (const float*)d_in[4];
  const float* bk = (const float*)d_in[5];
  const float* Wv = (const float*)d_in[6];
  const float* bv = (const float*)d_in[7];
  const float* Wo = (const float*)d_in[8];
  const float* bo = (const float*)d_in[9];
  float* out = (float*)d_out;

  const size_t NSD = (size_t)Bc * Sc * Dc;    // 8M elements
  __bf16* qb = (__bf16*)d_ws;
  __bf16* kb = qb + NSD;
  __bf16* vb = kb + NSD;
  __bf16* ob = vb + NSD;

  const int M = Bc * Sc;                      // 8192
  const int N = Dc, K = Dc;
  dim3 gGemm(M / 128, N / 128);               // 64 x 8
  dim3 blk(256);

  // QKV projections -> bf16 head-split layout
  gemm_bf16_wmma<false, true><<<gGemm, blk, 0, stream>>>(x, Wq, bq, qb, M, N, K);
  gemm_bf16_wmma<false, true><<<gGemm, blk, 0, stream>>>(x, Wk, bk, kb, M, N, K);
  gemm_bf16_wmma<false, true><<<gGemm, blk, 0, stream>>>(x, Wv, bv, vb, M, N, K);

  // flash attention -> bf16 [B,S,D]
  dim3 gAttn(Sc / 128, Bc * Hc);              // 16 x 64
  flash_attn<<<gAttn, blk, 0, stream>>>(qb, kb, vb, mask, ob);

  // output projection -> fp32 d_out
  gemm_bf16_wmma<true, false><<<gGemm, blk, 0, stream>>>(ob, Wo, bo, out, M, N, K);
}
